// HMM3Aligner_89129161326629
// MI455X (gfx1250) — matrix operand order repaired and checked by reading, not compiled, not run on hardware
//
#include <hip/hip_runtime.h>
#include <hip/hip_bf16.h>
#include <stdint.h>

#define B_   8
#define N_   512
#define M_   512
#define D_   256
#define NEGV (-1000000000.0f)

typedef _Float16 h8   __attribute__((ext_vector_type(8)));
typedef _Float16 v16h __attribute__((ext_vector_type(16)));
typedef float    v8f  __attribute__((ext_vector_type(8)));
typedef int      v4i  __attribute__((ext_vector_type(4)));

#if __has_builtin(__builtin_amdgcn_global_load_async_to_lds_b128)
#define HAVE_ASYNC_LDS 1
#else
#define HAVE_ASYNC_LDS 0
#endif

#if HAVE_ASYNC_LDS
#if __has_builtin(__builtin_amdgcn_s_wait_asynccnt)
#define WAIT_ASYNC(n) __builtin_amdgcn_s_wait_asynccnt(n)
#else
#define WAIT_ASYNC(n) asm volatile("s_wait_asynccnt %0" :: "i"(n) : "memory")
#endif
#else
#define WAIT_ASYNC(n)
#endif

// ---------------------------------------------------------------------------
// Pack: [B, L, D, 3] f32  ->  [B, 3, L, D] f16  (planar per-state, row-major K)
// ---------------------------------------------------------------------------
__global__ void pack_f16(const float* __restrict__ src, _Float16* __restrict__ dst) {
  const int idx = blockIdx.x * blockDim.x + threadIdx.x;   // over B*L*D
  const int total = B_ * N_ * D_;
  if (idx >= total) return;
  const int d = idx % D_;
  const int t = idx / D_;
  const int l = t % N_;
  const int b = t / N_;
  const float* s = src + ((size_t)(b * N_ + l) * D_ + d) * 3;
#pragma unroll
  for (int st = 0; st < 3; ++st)
    dst[((size_t)(b * 3 + st) * N_ + l) * D_ + d] = (_Float16)s[st];
}

// ---------------------------------------------------------------------------
// WMMA GEMM: per (batch, plane) computes C = X (512x256) * Y^T (256x512).
// Planes 0..2  -> theta[...,s]  (stride 3), planes 3..11 -> A[...,u,v] (stride 9).
// Block: 512 threads = 16 waves (4x4); block tile 128x128; wave tile 32x32
// (2x2 WMMA tiles -> 2x fragment reuse). K pipelined as 8 slabs of 32 through
// a ring of 3 LDS buffers with counted s_wait_asynccnt for overlap.
// ---------------------------------------------------------------------------
__global__ void __launch_bounds__(512)
gemm_wmma(const _Float16* __restrict__ Xz, const _Float16* __restrict__ Yz,
          const _Float16* __restrict__ Xg, const _Float16* __restrict__ Yg,
          float* __restrict__ theta, float* __restrict__ Aout)
{
  __shared__ _Float16 Xs[3][128][40];   // 40-half row stride: 80B, keeps 16B align
  __shared__ _Float16 Ys[3][128][40];   // total LDS: 3*2*128*40*2 = 61440 B

  const int bp = blockIdx.z;
  const int b  = bp / 12;
  const int p  = bp % 12;

  const _Float16* X; const _Float16* Y; float* out; int ss, sub;
  if (p < 3) {
    X = Xz + (size_t)(b * 3 + p) * N_ * D_;
    Y = Yz + (size_t)(b * 3 + p) * M_ * D_;
    out = theta; ss = 3; sub = p;
  } else {
    const int q = p - 3, u = q / 3, v = q % 3;
    X = Xg + (size_t)(b * 3 + u) * N_ * D_;
    Y = Yg + (size_t)(b * 3 + v) * M_ * D_;
    out = Aout; ss = 9; sub = q;
  }

  const int ti   = blockIdx.x * 128;
  const int tj   = blockIdx.y * 128;
  const int t    = threadIdx.x;
  const int lane = t & 31;
  const int w    = t >> 5;       // wave 0..15
  const int wi   = w >> 2;       // 0..3 (i-tile, 32 rows)
  const int wj   = w & 3;        // 0..3 (j-tile, 32 cols)

  // Staging: each thread moves one 16B chunk of X and one of Y per K-slab.
  const int srow = t >> 2;           // 0..127
  const int scol = (t & 3) * 8;      // 0,8,16,24
  const _Float16* gX = X + (size_t)(ti + srow) * D_ + scol;
  const _Float16* gY = Y + (size_t)(tj + srow) * D_ + scol;

#if HAVE_ASYNC_LDS
#define ISSUE_SLAB(s2)                                                          \
  do {                                                                          \
    __builtin_amdgcn_global_load_async_to_lds_b128(                             \
        (v4i*)(gX + (s2) * 32), (v4i*)&Xs[(s2) % 3][srow][scol], 0, 0);         \
    __builtin_amdgcn_global_load_async_to_lds_b128(                             \
        (v4i*)(gY + (s2) * 32), (v4i*)&Ys[(s2) % 3][srow][scol], 0, 0);         \
  } while (0)
#else
#define ISSUE_SLAB(s2)                                                          \
  do {                                                                          \
    *(uint4*)&Xs[(s2) % 3][srow][scol] = *(const uint4*)(gX + (s2) * 32);       \
    *(uint4*)&Ys[(s2) % 3][srow][scol] = *(const uint4*)(gY + (s2) * 32);       \
  } while (0)
#endif

  ISSUE_SLAB(0);
  ISSUE_SLAB(1);

  v8f acc00 = {}, acc01 = {}, acc10 = {}, acc11 = {};

  const int mrow = (lane & 15);
  const int koa  = (lane < 16) ? 0 : 8;    // A: K {0-7,16-23} / {8-15,24-31}
  const int kob  = (lane < 16) ? 0 : 8;    // B: K {0-7,16-23} / {8-15,24-31}

#pragma unroll
  for (int s = 0; s < 8; ++s) {
    // Wait until slab s (this wave's 2 oldest async loads) has landed, then
    // publish across the workgroup. Two slabs (4 loads) may stay in flight.
    if (s <= 5)      { WAIT_ASYNC(4); }
    else if (s == 6) { WAIT_ASYNC(2); }
    else             { WAIT_ASYNC(0); }
    __syncthreads();          // slab s visible; slab s-1 readers retired
    if (s + 2 < 8) ISSUE_SLAB(s + 2);   // safe: ring slot (s+2)%3 == (s-1)%3

    const int rs = s % 3;

    // A fragments (rows wi*32 + {0,16})
    const int ma0 = wi * 32 + mrow;
    h8 a0lo = *(const h8*)&Xs[rs][ma0][koa];
    h8 a0hi = *(const h8*)&Xs[rs][ma0][koa + 16];
    h8 a1lo = *(const h8*)&Xs[rs][ma0 + 16][koa];
    h8 a1hi = *(const h8*)&Xs[rs][ma0 + 16][koa + 16];
    v16h a0 = __builtin_shufflevector(a0lo, a0hi, 0,1,2,3,4,5,6,7,8,9,10,11,12,13,14,15);
    v16h a1 = __builtin_shufflevector(a1lo, a1hi, 0,1,2,3,4,5,6,7,8,9,10,11,12,13,14,15);

    // B fragments (cols wj*32 + {0,16})
    const int nb0 = wj * 32 + mrow;
    h8 b0lo = *(const h8*)&Ys[rs][nb0][kob];
    h8 b0hi = *(const h8*)&Ys[rs][nb0][kob + 16];
    h8 b1lo = *(const h8*)&Ys[rs][nb0 + 16][kob];
    h8 b1hi = *(const h8*)&Ys[rs][nb0 + 16][kob + 16];
    v16h b0 = __builtin_shufflevector(b0lo, b0hi, 0,1,2,3,4,5,6,7,8,9,10,11,12,13,14,15);
    v16h b1 = __builtin_shufflevector(b1lo, b1hi, 0,1,2,3,4,5,6,7,8,9,10,11,12,13,14,15);

    acc00 = __builtin_amdgcn_wmma_f32_16x16x32_f16(false, a0, false, b0, (short)0, acc00, false, false);
    acc01 = __builtin_amdgcn_wmma_f32_16x16x32_f16(false, a0, false, b1, (short)0, acc01, false, false);
    acc10 = __builtin_amdgcn_wmma_f32_16x16x32_f16(false, a1, false, b0, (short)0, acc10, false, false);
    acc11 = __builtin_amdgcn_wmma_f32_16x16x32_f16(false, a1, false, b1, (short)0, acc11, false, false);
  }
#undef ISSUE_SLAB

  // C/D layout: VGPR r -> (M=r, N=lane) lanes 0-15, (M=8+r, N=lane-16) lanes 16-31
  const int nbase = tj + wj * 32 + (lane & 15);
  const int mbase = ti + wi * 32 + ((lane & 16) ? 8 : 0);
#pragma unroll
  for (int r = 0; r < 8; ++r) {
    const size_t row0 = (size_t)b * N_ + (mbase + r);
    const size_t row1 = (size_t)b * N_ + (mbase + 16 + r);
    out[(row0 * M_ + nbase)      * ss + sub] = acc00[r];
    out[(row0 * M_ + nbase + 16) * ss + sub] = acc01[r];
    out[(row1 * M_ + nbase)      * ss + sub] = acc10[r];
    out[(row1 * M_ + nbase + 16) * ss + sub] = acc11[r];
  }
}

// ---------------------------------------------------------------------------
__device__ __forceinline__ float lse3(float a, float b, float c) {
  const float m = fmaxf(a, fmaxf(b, c));
  return m + __logf(__expf(a - m) + __expf(b - m) + __expf(c - m));
}

// ---------------------------------------------------------------------------
// Forward wavefront DP. One block per batch; thread j owns column j; diagonal
// d holds cell (d-j, j). Dependencies (i-1,j-1)@d-2, (i-1,j)@d-1, (i,j-1)@d-1
// -> three rotating LDS diagonal buffers, one barrier per diagonal.
// ---------------------------------------------------------------------------
__global__ void __launch_bounds__(512)
dp_forward(const float* __restrict__ theta, const float* __restrict__ Aw,
           float* __restrict__ Vw)
{
  __shared__ float buf[3][M_][3];
  const int b = blockIdx.x;
  const int j = threadIdx.x;
  const float* thb = theta + (size_t)b * N_ * M_ * 3;
  const float* Ab  = Aw    + (size_t)b * N_ * M_ * 9;
  float*       Vb  = Vw    + (size_t)b * N_ * M_ * 3;

  for (int d0 = 0; d0 < N_ + M_ - 1; ++d0) {
    const int i = d0 - j;
    if (i >= 0 && i < N_) {
      const int bc = d0 % 3, b1 = (d0 + 2) % 3, b2 = (d0 + 1) % 3;
      float pd0 = NEGV, pd1 = NEGV, pd2 = NEGV;   // V[i-1][j-1]
      float pu0 = NEGV, pu1 = NEGV, pu2 = NEGV;   // V[i-1][j]
      float pl0 = NEGV, pl1 = NEGV, pl2 = NEGV;   // V[i][j-1]
      if (i > 0 && j > 0) {
        pd0 = buf[b2][j-1][0]; pd1 = buf[b2][j-1][1]; pd2 = buf[b2][j-1][2];
      } else if (i == 0 && j == 0) {
        pd0 = 0.0f;                                // V0[0,0] = 0 (m-state start)
      }
      if (i > 0) { pu0 = buf[b1][j][0];   pu1 = buf[b1][j][1];   pu2 = buf[b1][j][2]; }
      if (j > 0) { pl0 = buf[b1][j-1][0]; pl1 = buf[b1][j-1][1]; pl2 = buf[b1][j-1][2]; }

      const float* th = thb + ((size_t)i * M_ + j) * 3;
      const float* Ac = Ab  + ((size_t)i * M_ + j) * 9;   // Ac[u*3+v]
      const float Vm = th[0] + lse3(pd0 + Ac[0], pd1 + Ac[3], pd2 + Ac[6]);
      const float Vx = th[1] + lse3(pu0 + Ac[1], pu1 + Ac[4], pu2 + Ac[7]);
      const float Vy = th[2] + lse3(pl0 + Ac[2], pl1 + Ac[5], pl2 + Ac[8]);

      buf[bc][j][0] = Vm; buf[bc][j][1] = Vx; buf[bc][j][2] = Vy;
      float* vp = Vb + ((size_t)i * M_ + j) * 3;
      vp[0] = Vm; vp[1] = Vx; vp[2] = Vy;
    }
    __syncthreads();
  }
}

// ---------------------------------------------------------------------------
// Backward wavefront: adjoint E[i,j,s] = dScore/dV[i,j,s]; grad wrt theta == E.
// E from successors (i+1,j+1)->m, (i+1,j)->x, (i,j+1)->y with softmax weights.
// ---------------------------------------------------------------------------
__global__ void __launch_bounds__(512)
dp_backward(const float* __restrict__ theta, const float* __restrict__ Aw,
            const float* __restrict__ Vw, float* __restrict__ out)
{
  __shared__ float Ebuf[3][M_][3];
  __shared__ float s_score;
  const int b = blockIdx.x;
  const int j = threadIdx.x;
  const float* thb = theta + (size_t)b * N_ * M_ * 3;
  const float* Ab  = Aw    + (size_t)b * N_ * M_ * 9;
  const float* Vb  = Vw    + (size_t)b * N_ * M_ * 3;
  float*       ob  = out   + (size_t)b * N_ * M_ * 3;

  if (j == 0) {
    const float* vt = Vb + ((size_t)(N_ - 1) * M_ + (M_ - 1)) * 3;
    s_score = lse3(vt[0], vt[1], vt[2]);
  }
  __syncthreads();
  const float score = s_score;

  for (int d0 = N_ + M_ - 2; d0 >= 0; --d0) {
    const int i = d0 - j;
    if (i >= 0 && i < N_) {
      const int bc = d0 % 3, p1 = (d0 + 1) % 3, p2 = (d0 + 2) % 3;
      const float* vc = Vb + ((size_t)i * M_ + j) * 3;
      float E0, E1, E2;
      if (i == N_ - 1 && j == M_ - 1) {
        E0 = __expf(vc[0] - score);
        E1 = __expf(vc[1] - score);
        E2 = __expf(vc[2] - score);
      } else {
        E0 = E1 = E2 = 0.0f;
        if (i + 1 < N_ && j + 1 < M_) {           // successor m-state @ d0+2
          const size_t c = (size_t)(i + 1) * M_ + (j + 1);
          const float e = Ebuf[p2][j + 1][0];
          const float k = thb[c * 3 + 0] - Vb[c * 3 + 0];
          const float* As = Ab + c * 9;
          E0 += e * __expf(vc[0] + As[0] + k);
          E1 += e * __expf(vc[1] + As[3] + k);
          E2 += e * __expf(vc[2] + As[6] + k);
        }
        if (i + 1 < N_) {                         // successor x-state @ d0+1
          const size_t c = (size_t)(i + 1) * M_ + j;
          const float e = Ebuf[p1][j][1];
          const float k = thb[c * 3 + 1] - Vb[c * 3 + 1];
          const float* As = Ab + c * 9;
          E0 += e * __expf(vc[0] + As[1] + k);
          E1 += e * __expf(vc[1] + As[4] + k);
          E2 += e * __expf(vc[2] + As[7] + k);
        }
        if (j + 1 < M_) {                         // successor y-state @ d0+1
          const size_t c = (size_t)i * M_ + (j + 1);
          const float e = Ebuf[p1][j + 1][2];
          const float k = thb[c * 3 + 2] - Vb[c * 3 + 2];
          const float* As = Ab + c * 9;
          E0 += e * __expf(vc[0] + As[2] + k);
          E1 += e * __expf(vc[1] + As[5] + k);
          E2 += e * __expf(vc[2] + As[8] + k);
        }
      }
      Ebuf[bc][j][0] = E0; Ebuf[bc][j][1] = E1; Ebuf[bc][j][2] = E2;
      float* op = ob + ((size_t)i * M_ + j) * 3;
      op[0] = E0; op[1] = E1; op[2] = E2;
    }
    __syncthreads();
  }
}

// ---------------------------------------------------------------------------
extern "C" void kernel_launch(void* const* d_in, const int* in_sizes, int n_in,
                              void* d_out, int out_size, void* d_ws, size_t ws_size,
                              hipStream_t stream)
{
  (void)in_sizes; (void)n_in; (void)out_size; (void)ws_size;
  const float* zx = (const float*)d_in[0];
  const float* zy = (const float*)d_in[1];
  const float* gx = (const float*)d_in[2];
  const float* gy = (const float*)d_in[3];

  char* w = (char*)d_ws;
  const size_t nTh = (size_t)B_ * N_ * M_ * 3;
  const size_t nA  = (size_t)B_ * N_ * M_ * 9;
  const size_t nPk = (size_t)B_ * 3 * N_ * D_;
  float*    theta = (float*)w;    w += nTh * sizeof(float);
  float*    Aw    = (float*)w;    w += nA  * sizeof(float);
  float*    Vw    = (float*)w;    w += nTh * sizeof(float);
  _Float16* Xz    = (_Float16*)w; w += nPk * sizeof(_Float16);
  _Float16* Yz    = (_Float16*)w; w += nPk * sizeof(_Float16);
  _Float16* Xg    = (_Float16*)w; w += nPk * sizeof(_Float16);
  _Float16* Yg    = (_Float16*)w; w += nPk * sizeof(_Float16);

  const int packBlocks = (B_ * N_ * D_ + 255) / 256;
  pack_f16<<<packBlocks, 256, 0, stream>>>(zx, Xz);
  pack_f16<<<packBlocks, 256, 0, stream>>>(zy, Yz);
  pack_f16<<<packBlocks, 256, 0, stream>>>(gx, Xg);
  pack_f16<<<packBlocks, 256, 0, stream>>>(gy, Yg);

  dim3 gg(N_ / 128, M_ / 128, B_ * 12);
  gemm_wmma<<<gg, 512, 0, stream>>>(Xz, Yz, Xg, Yg, theta, Aw);

  dp_forward<<<B_, M_, 0, stream>>>(theta, Aw, Vw);
  dp_backward<<<B_, M_, 0, stream>>>(theta, Aw, Vw, (float*)d_out);
}
